// PointNetSA_76596446756881
// MI455X (gfx1250) — matrix operand order repaired and checked by reading, not compiled, not run on hardware
//
#include <hip/hip_runtime.h>
#include <math.h>

typedef __attribute__((ext_vector_type(16))) _Float16 v16h;
typedef __attribute__((ext_vector_type(2)))  _Float16 h2;
typedef __attribute__((ext_vector_type(8)))  float    v8f;
typedef __attribute__((ext_vector_type(4)))  int      i4;

#define B_ 4
#define N_ 16384
#define S_ 1024
#define K_ 32
#define D_ 64
#define CIN 67
#define KP0 96          // padded input channels (3 k-chunks of 32)
#define C0 64
#define C1 64
#define C2 128
#define BN_EPS 1e-3f

// ---- workspace layout ----
// Weights stored TRANSPOSED (n-major, k-contiguous) so a WMMA B-fragment is
// two runs of 8 consecutive halfs (two ds_load_b128).
// Channel order of layer-0 input: [points(64), xyz(3), pad(29)].
#define WT0_OFF  0        // halfs: [64][96]  = 6144
#define WT1_OFF  6144     // halfs: [64][64]  = 4096
#define WT2_OFF  10240    // halfs: [128][64] = 8192
#define W_HALFS  18432    // total halfs = 36864 B
#define WS_BIAS  36864    // 256 floats (b0[64], b1[64], b2[128])
#define WS_KNN   40960    // 4*1024*32 ints

#if __has_builtin(__builtin_amdgcn_global_load_async_to_lds_b128)
#define PN_HAVE_ASYNC_LDS 1
typedef __attribute__((address_space(1))) i4* pn_gptr;
typedef __attribute__((address_space(3))) i4* pn_lptr;
#else
#define PN_HAVE_ASYNC_LDS 0
#endif

// ============================================================
// Kernel 0: fold BN scale into weights; emit transposed f16 weights + f32 bias
// ============================================================
__global__ void pn_prep(const float* w0, const float* g0, const float* b0,
                        const float* mm0, const float* mv0,
                        const float* w1, const float* g1, const float* b1,
                        const float* mm1, const float* mv1,
                        const float* w2, const float* g2, const float* b2,
                        const float* mm2, const float* mv2,
                        _Float16* wh, float* bias) {
    int tid = threadIdx.x;
    for (int c = tid; c < C0; c += blockDim.x) {
        float s = g0[c] * rsqrtf(mv0[c] + BN_EPS);
        bias[c] = b0[c] - mm0[c] * s;
    }
    for (int c = tid; c < C1; c += blockDim.x) {
        float s = g1[c] * rsqrtf(mv1[c] + BN_EPS);
        bias[64 + c] = b1[c] - mm1[c] * s;
    }
    for (int c = tid; c < C2; c += blockDim.x) {
        float s = g2[c] * rsqrtf(mv2[c] + BN_EPS);
        bias[128 + c] = b2[c] - mm2[c] * s;
    }
    // W0^T: [n=64][k=96]; channel k: k<64 -> points row (3+k), 64..66 -> xyz row (k-64)
    for (int i = tid; i < C0 * KP0; i += blockDim.x) {
        int n = i / KP0, k = i % KP0;
        float s = g0[n] * rsqrtf(mv0[n] + BN_EPS);
        float v = 0.0f;
        if (k < 64)       v = w0[(3 + k) * C0 + n];
        else if (k < CIN) v = w0[(k - 64) * C0 + n];
        wh[WT0_OFF + i] = (_Float16)(v * s);
    }
    // W1^T: [n=64][k=64]
    for (int i = tid; i < C1 * C0; i += blockDim.x) {
        int n = i / C0, k = i % C0;
        float s = g1[n] * rsqrtf(mv1[n] + BN_EPS);
        wh[WT1_OFF + i] = (_Float16)(w1[k * C1 + n] * s);
    }
    // W2^T: [n=128][k=64]
    for (int i = tid; i < C2 * C1; i += blockDim.x) {
        int n = i / C1, k = i % C1;
        float s = g2[n] * rsqrtf(mv2[n] + BN_EPS);
        wh[WT2_OFF + i] = (_Float16)(w2[k * C2 + n] * s);
    }
}

// ============================================================
// Kernel 1: farthest point sampling; writes new_xyz directly.
// ============================================================
__global__ void pn_fps(const float* __restrict__ xyz, float* __restrict__ new_xyz) {
    __shared__ float rv[1024];
    __shared__ int   ri[1024];
    __shared__ float cent[3];

    const int tid = threadIdx.x;
    const int b = blockIdx.x;
    const float* base = xyz + (size_t)b * N_ * 3;

    float px[16], py[16], pz[16], dist[16];
#pragma unroll
    for (int j = 0; j < 16; ++j) {
        int i = j * 1024 + tid;
        px[j] = base[i * 3 + 0];
        py[j] = base[i * 3 + 1];
        pz[j] = base[i * 3 + 2];
        dist[j] = 1e10f;
    }

    int far = 0;
    for (int it = 0; it < S_; ++it) {
        if (tid == 0) {
            const float* c = base + (size_t)far * 3;
            cent[0] = c[0]; cent[1] = c[1]; cent[2] = c[2];
        }
        __syncthreads();
        if (tid < 3) new_xyz[((size_t)b * S_ + it) * 3 + tid] = cent[tid];
        float cx = cent[0], cy = cent[1], cz = cent[2];

        float lv = -1.0f; int li = 0;
#pragma unroll
        for (int j = 0; j < 16; ++j) {
            float dx = px[j] - cx, dy = py[j] - cy, dz = pz[j] - cz;
            float d = dx * dx + dy * dy + dz * dz;
            dist[j] = fminf(dist[j], d);
            if (dist[j] > lv) { lv = dist[j]; li = j * 1024 + tid; }
        }
        rv[tid] = lv; ri[tid] = li;
        __syncthreads();
        for (int off = 512; off > 0; off >>= 1) {
            if (tid < off) {
                if (rv[tid + off] > rv[tid]) { rv[tid] = rv[tid + off]; ri[tid] = ri[tid + off]; }
            }
            __syncthreads();
        }
        far = ri[0];
    }
}

// ============================================================
// Kernel 2: 32-NN per query (1 thread/query, LDS-staged candidates).
// ============================================================
__global__ void pn_knn(const float* __restrict__ xyz, const float* __restrict__ new_xyz,
                       int* __restrict__ knn) {
    __shared__ float sc[2048 * 3];
    const int tid = threadIdx.x;
    const int b = blockIdx.x >> 2;
    const int q = ((blockIdx.x & 3) << 8) + tid;
    const size_t qg = (size_t)b * S_ + q;

    const float qx = new_xyz[qg * 3 + 0];
    const float qy = new_xyz[qg * 3 + 1];
    const float qz = new_xyz[qg * 3 + 2];

    float bd[K_]; int bi[K_];
#pragma unroll
    for (int j = 0; j < K_; ++j) { bd[j] = 3e30f; bi[j] = 0; }

    const float* base = xyz + (size_t)b * N_ * 3;
    for (int ch = 0; ch < N_ / 2048; ++ch) {
        for (int i = tid; i < 2048 * 3; i += 256) sc[i] = base[(size_t)ch * 2048 * 3 + i];
        __syncthreads();
        const int c0 = ch * 2048;
        for (int i = 0; i < 2048; ++i) {
            float dx = sc[i * 3 + 0] - qx;
            float dy = sc[i * 3 + 1] - qy;
            float dz = sc[i * 3 + 2] - qz;
            float dd = dx * dx + dy * dy + dz * dz;
            if (dd < bd[K_ - 1]) {
                int j = K_ - 1;
                while (j > 0 && bd[j - 1] > dd) { bd[j] = bd[j - 1]; bi[j] = bi[j - 1]; --j; }
                bd[j] = dd; bi[j] = c0 + i;
            }
        }
        __syncthreads();
    }
    for (int j = 0; j < K_; ++j) knn[qg * K_ + j] = bi[j];
}

// ============================================================
// WMMA fragment loaders per CDNA5 ISA 16-bit operand layouts (§7.12.2)
// A 16x32: lanes 0-15 -> K {0..7,16..23}; lanes 16-31 -> K {8..15,24..31}
// B 32x16: lanes 0-15 -> K {0..7,16..23}; lanes 16-31 -> K {8..15,24..31} (k-contiguous in W^T)
// C/D:     vgpr r -> M = r + 8*(lane>=16), N = lane&15
// ============================================================
__device__ __forceinline__ v16h loadA(const _Float16* buf, int stride, int mtile,
                                      int kc, int lane) {
    v16h a;
    const _Float16* p = buf + ((lane & 15) + mtile * 16) * stride + kc * 32 + ((lane >= 16) ? 8 : 0);
#pragma unroll
    for (int h = 0; h < 16; ++h) a[h] = p[(h < 8) ? h : (h + 8)];
    return a;
}

__device__ __forceinline__ v16h loadB(const _Float16* wT, int kstride, int ntile,
                                      int kc, int lane) {
    v16h bf;
    const _Float16* p = wT + (ntile * 16 + (lane & 15)) * kstride + kc * 32 + ((lane >= 16) ? 8 : 0);
#pragma unroll
    for (int h = 0; h < 16; ++h) bf[h] = p[(h < 8) ? h : (h + 8)];   // two 8-half runs -> 2x ds_load_b128
    return bf;
}

template <int KCH, int NCOLS, int KSTR>
__device__ __forceinline__ void layer_fwd(const _Float16* xin, int si,
                                          const _Float16* wT, const float* bias,
                                          _Float16* xout, int so, int lane) {
    const int r0 = (lane >= 16) ? 8 : 0;
    const int nl = lane & 15;
#pragma unroll
    for (int mtile = 0; mtile < 2; ++mtile) {
        v16h a[KCH];
#pragma unroll
        for (int kc = 0; kc < KCH; ++kc) a[kc] = loadA(xin, si, mtile, kc, lane);
#pragma unroll 1
        for (int ntile = 0; ntile < NCOLS / 16; ++ntile) {
            v8f c = {};
#pragma unroll
            for (int kc = 0; kc < KCH; ++kc) {
                v16h bb = loadB(wT, KSTR, ntile, kc, lane);
                c = __builtin_amdgcn_wmma_f32_16x16x32_f16(false, a[kc], false, bb,
                                                           (short)0, c, false, false);
            }
            const float bv = bias[ntile * 16 + nl];
            const int col = nl + ntile * 16;
#pragma unroll
            for (int r = 0; r < 8; ++r) {
                float v = fmaxf(c[r] + bv, 0.0f);
                xout[(mtile * 16 + r0 + r) * so + col] = (_Float16)v;
            }
        }
    }
}

// ============================================================
// Kernel 3: gather + 3-layer WMMA MLP + max-pool. 1 wave = 1 group.
// grid = 2048, block = 64 (2 waves). Static LDS = 57344 B.
// ============================================================
__global__ void pn_mlp(const float* __restrict__ xyz, const float* __restrict__ points,
                       const _Float16* __restrict__ wts, const float* __restrict__ bias,
                       const int* __restrict__ knn, const float* __restrict__ new_xyz,
                       float* __restrict__ out_points) {
    __shared__ _Float16 sW[W_HALFS];      // 36864 B (W0^T | W1^T | W2^T)
    __shared__ _Float16 sX0[2][32 * KP0]; // 12288 B
    __shared__ _Float16 sX1[2][32 * C0];  //  8192 B

    const int tid = threadIdx.x;
    // ---- stage weights: async global->LDS (ASYNCcnt) if available ----
#if PN_HAVE_ASYNC_LDS
    {
        for (int i = tid; i < W_HALFS / 8; i += 64) {   // 16B per lane per op
            __builtin_amdgcn_global_load_async_to_lds_b128(
                (pn_gptr)(wts + i * 8), (pn_lptr)(sW + i * 8), 0, 0);
        }
        asm volatile("s_wait_asynccnt 0" ::: "memory");
    }
#else
    {
        const uint4* gsrc = (const uint4*)wts;
        uint4* ldst = (uint4*)sW;
        for (int i = tid; i < W_HALFS / 8; i += 64) ldst[i] = gsrc[i];
    }
#endif
    __syncthreads();

    const _Float16* sW0 = sW + WT0_OFF;
    const _Float16* sW1 = sW + WT1_OFF;
    const _Float16* sW2 = sW + WT2_OFF;

    const int wave = tid >> 5, lane = tid & 31;
    const int g = blockIdx.x * 2 + wave;  // 0..4095
    const int b = g >> 10;
    _Float16* x0 = sX0[wave];
    _Float16* x1 = sX1[wave];

    // ---- gather: lane = sample k; channels = [points(64), xyz(3), pad] ----
    {
        const int pid = knn[(size_t)g * K_ + lane];
        const float* F = points + ((size_t)b * N_ + pid) * D_;
        _Float16* row = x0 + lane * KP0;
#pragma unroll 8
        for (int c = 0; c < D_; c += 2)
            *(h2*)(row + c) = (h2){(_Float16)F[c], (_Float16)F[c + 1]};
        const float* P = xyz + ((size_t)b * N_ + pid) * 3;
        const float* Q = new_xyz + (size_t)g * 3;
        row[64] = (_Float16)(P[0] - Q[0]);
        row[65] = (_Float16)(P[1] - Q[1]);
        *(h2*)(row + 66) = (h2){(_Float16)(P[2] - Q[2]), (_Float16)0.0f};
#pragma unroll
        for (int c = 68; c < KP0; c += 2) *(h2*)(row + c) = (h2){(_Float16)0.0f, (_Float16)0.0f};
    }
    __syncthreads();

    layer_fwd<3, C0, KP0>(x0, KP0, sW0, bias,      x1, C0, lane);  // 96 -> 64
    __syncthreads();
    layer_fwd<2, C1, C0>(x1, C0,  sW1, bias + 64,  x0, C1, lane);  // 64 -> 64
    __syncthreads();

    // ---- layer 2 (64 -> 128) fused with max-pool over K=32 rows ----
    {
        v16h a[2][2];
#pragma unroll
        for (int mtile = 0; mtile < 2; ++mtile)
#pragma unroll
            for (int kc = 0; kc < 2; ++kc) a[mtile][kc] = loadA(x0, C1, mtile, kc, lane);

        const int nl = lane & 15;
#pragma unroll 1
        for (int ntile = 0; ntile < C2 / 16; ++ntile) {
            const float bv = bias[128 + ntile * 16 + nl];
            float m = -3e30f;
#pragma unroll
            for (int mtile = 0; mtile < 2; ++mtile) {
                v8f c = {};
#pragma unroll
                for (int kc = 0; kc < 2; ++kc) {
                    v16h bb = loadB(sW2, C1, ntile, kc, lane);
                    c = __builtin_amdgcn_wmma_f32_16x16x32_f16(false, a[mtile][kc], false, bb,
                                                               (short)0, c, false, false);
                }
#pragma unroll
                for (int r = 0; r < 8; ++r) m = fmaxf(m, fmaxf(c[r] + bv, 0.0f));
            }
            m = fmaxf(m, __shfl_xor(m, 16, 32));   // combine M halves
            if (lane < 16)
                out_points[(size_t)g * C2 + ntile * 16 + lane] = m;
        }
    }
}

// ============================================================
extern "C" void kernel_launch(void* const* d_in, const int* in_sizes, int n_in,
                              void* d_out, int out_size, void* d_ws, size_t ws_size,
                              hipStream_t stream) {
    const float* xyz = (const float*)d_in[0];
    const float* pts = (const float*)d_in[1];
    const float* w0  = (const float*)d_in[2];
    const float* g0  = (const float*)d_in[3];
    const float* b0  = (const float*)d_in[4];
    const float* mm0 = (const float*)d_in[5];
    const float* mv0 = (const float*)d_in[6];
    const float* w1  = (const float*)d_in[7];
    const float* g1  = (const float*)d_in[8];
    const float* b1  = (const float*)d_in[9];
    const float* mm1 = (const float*)d_in[10];
    const float* mv1 = (const float*)d_in[11];
    const float* w2  = (const float*)d_in[12];
    const float* g2  = (const float*)d_in[13];
    const float* b2  = (const float*)d_in[14];
    const float* mm2 = (const float*)d_in[15];
    const float* mv2 = (const float*)d_in[16];

    float* out = (float*)d_out;
    float* new_xyz = out;                      // [4,1024,3]
    float* new_points = out + B_ * S_ * 3;     // [4,1024,128]

    char* ws = (char*)d_ws;
    _Float16* wh = (_Float16*)ws;
    float* bias  = (float*)(ws + WS_BIAS);
    int* knn     = (int*)(ws + WS_KNN);

    pn_prep<<<1, 256, 0, stream>>>(w0, g0, b0, mm0, mv0,
                                   w1, g1, b1, mm1, mv1,
                                   w2, g2, b2, mm2, mv2, wh, bias);
    pn_fps<<<B_, 1024, 0, stream>>>(xyz, new_xyz);
    pn_knn<<<B_ * 4, 256, 0, stream>>>(xyz, new_xyz, knn);
    pn_mlp<<<(B_ * S_) / 2, 64, 0, stream>>>(xyz, pts, wh, bias, knn, new_xyz, new_points);
}
